// SimpleMoEForExport_61031485276534
// MI455X (gfx1250) — matrix operand order repaired and checked by reading, not compiled, not run on hardware
//
#include <hip/hip_runtime.h>
#include <hip/hip_bf16.h>

// ---------------------------------------------------------------------------
// MoE (top-2 of 8, SwiGLU) for MI455X / gfx1250.
// bf16 WMMA GEMMs (fp32 accum), token compaction, double-buffered K=64 LDS
// tiles fed by GLOBAL_LOAD_ASYNC_TO_LDS, software-pipelined, wave32.
// ---------------------------------------------------------------------------

typedef __bf16 bf16;
typedef bf16  bf16x8  __attribute__((ext_vector_type(8)));
typedef bf16  bf16x16 __attribute__((ext_vector_type(16)));
typedef float floatx8 __attribute__((ext_vector_type(8)));

static constexpr int T = 8192;   // tokens
static constexpr int H = 2048;   // hidden
static constexpr int I = 4096;   // intermediate
static constexpr int E = 8;      // experts

// LDS row stride (halves) for K=64 tiles: 64 + 8 pad -> 144B, conflict-free
static constexpr int LP = 72;

// ---------------------------------------------------------------------------
// helpers
// ---------------------------------------------------------------------------
__device__ __forceinline__ bf16x16 join8(bf16x8 lo, bf16x8 hi) {
  return __builtin_shufflevector(lo, hi, 0,1,2,3,4,5,6,7,8,9,10,11,12,13,14,15);
}

// A fragment: 16x32 bf16 tile, lane m = lane%16.
// lanes 0-15: K0..15, lanes 16-31: K16..31 (16 contiguous halves).
__device__ __forceinline__ bf16x16 load_a_frag(const bf16* row, int lhalf) {
  const int c0 = lhalf * 16;
  bf16x8 lo = *(const bf16x8*)(row + c0);
  bf16x8 hi = *(const bf16x8*)(row + c0 + 8);
  return join8(lo, hi);
}

// B fragment: 32x16 bf16 tile (transposed weights: row n along K).
// lanes 0-15: K0..15, lanes 16-31: K16..31 (16 contiguous halves).
__device__ __forceinline__ bf16x16 load_b_frag(const bf16* row, int lhalf) {
  const int c0 = lhalf * 16;
  bf16x8 lo = *(const bf16x8*)(row + c0);
  bf16x8 hi = *(const bf16x8*)(row + c0 + 8);
  return join8(lo, hi);
}

__device__ __forceinline__ bf16x8 cvt8(float4 a, float4 b) {
  bf16x8 p;
  p[0] = (bf16)a.x; p[1] = (bf16)a.y; p[2] = (bf16)a.z; p[3] = (bf16)a.w;
  p[4] = (bf16)b.x; p[5] = (bf16)b.y; p[6] = (bf16)b.z; p[7] = (bf16)b.w;
  return p;
}

// CDNA5 async copy: global -> LDS, 16 bytes per lane, tracked by ASYNCcnt.
__device__ __forceinline__ void async_ld16(unsigned lds_addr, const void* gaddr) {
  asm volatile("global_load_async_to_lds_b128 %0, %1, off"
               :: "v"(lds_addr), "v"((unsigned long long)(uintptr_t)gaddr)
               : "memory");
}
__device__ __forceinline__ void wait_async0() {
  asm volatile("s_wait_asynccnt 0x0" ::: "memory");
}

__device__ __forceinline__ unsigned lds_off(const void* p) {
  return (unsigned)(uintptr_t)p;  // generic LDS ptr: low 32 bits = LDS offset
}

// copy 64B (32 halves) global -> LDS, async
__device__ __forceinline__ void async_copy64(const bf16* g, unsigned l) {
  async_ld16(l,      g);
  async_ld16(l + 16, g + 8);
  async_ld16(l + 32, g + 16);
  async_ld16(l + 48, g + 24);
}

// ---------------------------------------------------------------------------
// small kernels
// ---------------------------------------------------------------------------
__global__ void zero_out_kernel(float* __restrict__ out, int n4) {
  int i = blockIdx.x * blockDim.x + threadIdx.x;
  if (i < n4) ((float4*)out)[i] = make_float4(0.f, 0.f, 0.f, 0.f);
}

__global__ void zero_cnt_kernel(int* __restrict__ cnt) {
  if (threadIdx.x < E) cnt[threadIdx.x] = 0;
}

__global__ void cvt_x_kernel(const float* __restrict__ src, bf16* __restrict__ dst) {
  size_t i = ((size_t)blockIdx.x * blockDim.x + threadIdx.x) * 8;
  float4 v0 = *(const float4*)(src + i);
  float4 v1 = *(const float4*)(src + i + 4);
  *(bf16x8*)(dst + i) = cvt8(v0, v1);
}

// ---------------------------------------------------------------------------
// router: logits -> softmax -> top2 -> compacted per-expert token lists
// one wave (32 lanes) per token
// ---------------------------------------------------------------------------
__global__ __launch_bounds__(256) void router_kernel(
    const float* __restrict__ x, const float* __restrict__ gw,
    int* __restrict__ cnt, int* __restrict__ tok_idx, float* __restrict__ tok_coef) {
  const int wv = threadIdx.x >> 5;
  const int lane = threadIdx.x & 31;
  const int t = blockIdx.x * 8 + wv;

  float acc[E];
#pragma unroll
  for (int e = 0; e < E; ++e) acc[e] = 0.f;

  const float* xrow = x + (size_t)t * H;
  for (int h = lane; h < H; h += 32) {
    float xv = xrow[h];
#pragma unroll
    for (int e = 0; e < E; ++e) acc[e] += xv * gw[e * H + h];
  }
#pragma unroll
  for (int e = 0; e < E; ++e) {
#pragma unroll
    for (int off = 16; off > 0; off >>= 1) acc[e] += __shfl_xor(acc[e], off, 32);
  }

  if (lane == 0) {
    float m = acc[0];
#pragma unroll
    for (int e = 1; e < E; ++e) m = fmaxf(m, acc[e]);
    float p[E];
    float s = 0.f;
#pragma unroll
    for (int e = 0; e < E; ++e) { p[e] = __expf(acc[e] - m); s += p[e]; }
    const float inv = 1.f / s;

    int i1 = 0; float b1 = p[0];
#pragma unroll
    for (int e = 1; e < E; ++e) if (p[e] > b1) { b1 = p[e]; i1 = e; }
    int i2 = -1; float b2 = -1.f;
#pragma unroll
    for (int e = 0; e < E; ++e) if (e != i1 && p[e] > b2) { b2 = p[e]; i2 = e; }

    int s1 = atomicAdd(cnt + i1, 1);
    tok_idx[i1 * T + s1] = t;
    tok_coef[i1 * T + s1] = b1 * inv;
    int s2 = atomicAdd(cnt + i2, 1);
    tok_idx[i2 * T + s2] = t;
    tok_coef[i2 * T + s2] = b2 * inv;
  }
}

// ---------------------------------------------------------------------------
// GEMM1: h[slot, i] = silu(x.w1^T) * (x.w3^T) for compacted tokens of expert e
// WG tile: 128 slots x 64 inter, K staged 64 wide, double buffered.
// 8 waves, 2x2 WMMA tiles each, 16 WMMAs per wave per barrier.
// ---------------------------------------------------------------------------
__global__ __launch_bounds__(256) void gemm1_kernel(
    const bf16* __restrict__ xb, const float* __restrict__ w1f,
    const float* __restrict__ w3f, bf16* __restrict__ hb,
    const int* __restrict__ cnts, const int* __restrict__ tok_idx, int e) {
  const int cnt = cnts[e];
  const int m0 = blockIdx.x * 128;
  if (m0 >= cnt) return;
  const int i0 = blockIdx.y * 64;
  const int* idx = tok_idx + e * T;

  __shared__ __align__(16) bf16 sx[2][128][LP];
  __shared__ __align__(16) bf16 sw1[2][64][LP];
  __shared__ __align__(16) bf16 sw3[2][64][LP];

  const int tid = threadIdx.x;
  // x staging: 128 rows x 2 chunks of 32 halves (gathered rows, bf16, async)
  const int xr = tid >> 1, xc = (tid & 1) * 32;
  const int row = m0 + xr;
  const int tok = (row < cnt) ? idx[row] : 0;
  const bf16* xsrc = xb + (size_t)tok * H + xc;
  // weight staging: 64 rows x 4 chunks of 16 floats (fp32 -> bf16 on the fly)
  const int wr = tid >> 2, wc = (tid & 3) * 16;
  const float* w1src = w1f + ((size_t)e * I + i0 + wr) * H + wc;
  const float* w3src = w3f + ((size_t)e * I + i0 + wr) * H + wc;

  const int wv = tid >> 5, lane = tid & 31;
  const int m_off = (wv & 3) * 32, n_off = (wv >> 2) * 32;
  const int lrow = lane & 15, lhalf = lane >> 4;

  const floatx8 z = {0.f, 0.f, 0.f, 0.f, 0.f, 0.f, 0.f, 0.f};
  floatx8 accg[2][2], accu[2][2];
#pragma unroll
  for (int mi = 0; mi < 2; ++mi)
#pragma unroll
    for (int ni = 0; ni < 2; ++ni) { accg[mi][ni] = z; accu[mi][ni] = z; }

  float4 r1[4], r3[4];

  auto issue_x = [&](int b, int k) {
    async_copy64(xsrc + k, lds_off(&sx[b][xr][xc]));
  };
  auto load_w = [&](int k) {
#pragma unroll
    for (int q = 0; q < 4; ++q) {
      r1[q] = *(const float4*)(w1src + k + q * 4);
      r3[q] = *(const float4*)(w3src + k + q * 4);
    }
  };
  auto store_w = [&](int b) {
    *(bf16x8*)&sw1[b][wr][wc]     = cvt8(r1[0], r1[1]);
    *(bf16x8*)&sw1[b][wr][wc + 8] = cvt8(r1[2], r1[3]);
    *(bf16x8*)&sw3[b][wr][wc]     = cvt8(r3[0], r3[1]);
    *(bf16x8*)&sw3[b][wr][wc + 8] = cvt8(r3[2], r3[3]);
  };
  auto compute = [&](int b) {
#pragma unroll
    for (int kk = 0; kk < 2; ++kk) {
      bf16x16 a[2], b1[2], b3[2];
#pragma unroll
      for (int mi = 0; mi < 2; ++mi)
        a[mi] = load_a_frag(&sx[b][m_off + mi * 16 + lrow][kk * 32], lhalf);
#pragma unroll
      for (int ni = 0; ni < 2; ++ni) {
        b1[ni] = load_b_frag(&sw1[b][n_off + ni * 16 + lrow][kk * 32], lhalf);
        b3[ni] = load_b_frag(&sw3[b][n_off + ni * 16 + lrow][kk * 32], lhalf);
      }
#pragma unroll
      for (int mi = 0; mi < 2; ++mi)
#pragma unroll
        for (int ni = 0; ni < 2; ++ni) {
          accg[mi][ni] = __builtin_amdgcn_wmma_f32_16x16x32_bf16(
              false, a[mi], false, b1[ni], (short)0, accg[mi][ni], false, false);
          accu[mi][ni] = __builtin_amdgcn_wmma_f32_16x16x32_bf16(
              false, a[mi], false, b3[ni], (short)0, accu[mi][ni], false, false);
        }
    }
  };

  // prologue: stage slice 0 into buffer 0
  issue_x(0, 0);
  load_w(0);
  store_w(0);
  wait_async0();
  __syncthreads();

  int buf = 0;
  // steady state: stage slice k while computing slice k-64 (straight-line body)
  for (int k = 64; k < H; k += 64) {
    issue_x(buf ^ 1, k);
    load_w(k);
    __builtin_prefetch(w1src + k + 256, 0, 0);
    __builtin_prefetch(w3src + k + 256, 0, 0);
    compute(buf);
    store_w(buf ^ 1);
    wait_async0();
    __syncthreads();
    buf ^= 1;
  }
  // peeled last slice
  compute(buf);

  // epilogue: h = silu(g) * u, store bf16 (slot-major)
#pragma unroll
  for (int mi = 0; mi < 2; ++mi)
#pragma unroll
    for (int ni = 0; ni < 2; ++ni)
#pragma unroll
      for (int j = 0; j < 8; ++j) {
        float g = accg[mi][ni][j];
        float u = accu[mi][ni][j];
        float sg = g * __builtin_amdgcn_rcpf(1.0f + __expf(-g));
        int mrow = m0 + m_off + mi * 16 + lhalf * 8 + j;
        int ncol = i0 + n_off + ni * 16 + lrow;
        hb[(size_t)mrow * I + ncol] = (bf16)(sg * u);
      }
}

// ---------------------------------------------------------------------------
// GEMM2: out[tok, :] += coef * (h . w2^T). WG tile 128 slots x 64 hidden.
// Experts run sequentially on the stream -> race-free accumulate.
// ---------------------------------------------------------------------------
__global__ __launch_bounds__(256) void gemm2_kernel(
    const bf16* __restrict__ hb, const float* __restrict__ w2f,
    float* __restrict__ outp, const int* __restrict__ cnts,
    const int* __restrict__ tok_idx, const float* __restrict__ tok_coef, int e) {
  const int cnt = cnts[e];
  const int m0 = blockIdx.x * 128;
  if (m0 >= cnt) return;
  const int h0 = blockIdx.y * 64;
  const int* idx = tok_idx + e * T;
  const float* coef = tok_coef + e * T;

  __shared__ __align__(16) bf16 sh[2][128][LP];
  __shared__ __align__(16) bf16 sw[2][64][LP];

  const int tid = threadIdx.x;
  const int xr = tid >> 1, xc = (tid & 1) * 32;
  const bf16* hsrc = hb + (size_t)(m0 + xr) * I + xc;
  const int wr = tid >> 2, wc = (tid & 3) * 16;
  const float* wsrc = w2f + ((size_t)e * H + h0 + wr) * I + wc;

  const int wv = tid >> 5, lane = tid & 31;
  const int m_off = (wv & 3) * 32, n_off = (wv >> 2) * 32;
  const int lrow = lane & 15, lhalf = lane >> 4;

  const floatx8 z = {0.f, 0.f, 0.f, 0.f, 0.f, 0.f, 0.f, 0.f};
  floatx8 acc[2][2];
#pragma unroll
  for (int mi = 0; mi < 2; ++mi)
#pragma unroll
    for (int ni = 0; ni < 2; ++ni) acc[mi][ni] = z;

  float4 rw[4];

  auto issue_h = [&](int b, int k) {
    async_copy64(hsrc + k, lds_off(&sh[b][xr][xc]));
  };
  auto load_w = [&](int k) {
#pragma unroll
    for (int q = 0; q < 4; ++q) rw[q] = *(const float4*)(wsrc + k + q * 4);
  };
  auto store_w = [&](int b) {
    *(bf16x8*)&sw[b][wr][wc]     = cvt8(rw[0], rw[1]);
    *(bf16x8*)&sw[b][wr][wc + 8] = cvt8(rw[2], rw[3]);
  };
  auto compute = [&](int b) {
#pragma unroll
    for (int kk = 0; kk < 2; ++kk) {
      bf16x16 a[2], bfr[2];
#pragma unroll
      for (int mi = 0; mi < 2; ++mi)
        a[mi] = load_a_frag(&sh[b][m_off + mi * 16 + lrow][kk * 32], lhalf);
#pragma unroll
      for (int ni = 0; ni < 2; ++ni)
        bfr[ni] = load_b_frag(&sw[b][n_off + ni * 16 + lrow][kk * 32], lhalf);
#pragma unroll
      for (int mi = 0; mi < 2; ++mi)
#pragma unroll
        for (int ni = 0; ni < 2; ++ni)
          acc[mi][ni] = __builtin_amdgcn_wmma_f32_16x16x32_bf16(
              false, a[mi], false, bfr[ni], (short)0, acc[mi][ni], false, false);
    }
  };

  // prologue
  issue_h(0, 0);
  load_w(0);
  store_w(0);
  wait_async0();
  __syncthreads();

  int buf = 0;
  for (int k = 64; k < I; k += 64) {
    issue_h(buf ^ 1, k);
    load_w(k);
    __builtin_prefetch(wsrc + k + 256, 0, 0);
    compute(buf);
    store_w(buf ^ 1);
    wait_async0();
    __syncthreads();
    buf ^= 1;
  }
  compute(buf);

  // epilogue: scale by routing coef, scatter-accumulate to the token's row
#pragma unroll
  for (int mi = 0; mi < 2; ++mi)
#pragma unroll
    for (int j = 0; j < 8; ++j) {
      int s = m0 + m_off + mi * 16 + lhalf * 8 + j;
      if (s < cnt) {
        int t = idx[s];
        float c = coef[s];
        float* o = outp + (size_t)t * H + h0 + n_off + lrow;
        o[0]  += c * acc[mi][0][j];
        o[16] += c * acc[mi][1][j];
      }
    }
}

// ---------------------------------------------------------------------------
// launch
// ---------------------------------------------------------------------------
extern "C" void kernel_launch(void* const* d_in, const int* in_sizes, int n_in,
                              void* d_out, int out_size, void* d_ws, size_t ws_size,
                              hipStream_t stream) {
  (void)in_sizes; (void)n_in; (void)out_size; (void)ws_size;

  const float* x  = (const float*)d_in[0];
  const float* gw = (const float*)d_in[1];
  const float* w1 = (const float*)d_in[2];
  const float* w2 = (const float*)d_in[3];
  const float* w3 = (const float*)d_in[4];
  float* out = (float*)d_out;

  // workspace layout (bytes): xb[T*H bf16] | hb[T*I bf16] | idx | coef | cnt
  char* ws = (char*)d_ws;
  bf16*  xb   = (bf16*)(ws);                                   // 33,554,432
  bf16*  hb   = (bf16*)(ws + 33554432);                        // 67,108,864
  int*   idx  = (int*)(ws + 33554432 + 67108864);              //    262,144
  float* coef = (float*)(ws + 100925440);                      //    262,144
  int*   cnt  = (int*)(ws + 101187584);                        //         32

  zero_cnt_kernel<<<1, 32, 0, stream>>>(cnt);
  zero_out_kernel<<<(T * H / 4 + 255) / 256, 256, 0, stream>>>(out, T * H / 4);
  cvt_x_kernel<<<T * H / 8 / 256, 256, 0, stream>>>(x, xb);
  router_kernel<<<T / 8, 256, 0, stream>>>(x, gw, cnt, idx, coef);

  for (int e = 0; e < E; ++e) {
    gemm1_kernel<<<dim3(T / 128, I / 64), 256, 0, stream>>>(xb, w1, w3, hb, cnt, idx, e);
    gemm2_kernel<<<dim3(T / 128, H / 64), 256, 0, stream>>>(hb, w2, out, cnt, idx, coef, e);
  }
}
